// GraphDiscriminatorPointNet_68101001445996
// MI455X (gfx1250) — matrix operand order repaired and checked by reading, not compiled, not run on hardware
//
#include <hip/hip_runtime.h>
#include <hip/hip_bf16.h>

typedef __attribute__((ext_vector_type(16))) _Float16 v16h;
typedef __attribute__((ext_vector_type(8)))  float    v8f;

#define USE_ASYNC 1

// ---------------------------------------------------------------------------
// Pack f32 weight matrix W[K][NCOL] into f16 WMMA B-fragment order.
// For tile (nt, kt): block of 512 halves; lane L holds
//   b[j] = W[kt*32 + (L>>4)*8 + (j&7) + (j>>3)*16][nt*16 + (L&15)]
// K rows padded to KP (multiple of 32) with zeros.
// ---------------------------------------------------------------------------
__global__ void pack_weights_kernel(const float* __restrict__ src,
                                    _Float16* __restrict__ dst,
                                    int K, int NCOL, int KP) {
    int total = KP * NCOL;
    int idx = blockIdx.x * blockDim.x + threadIdx.x;
    if (idx >= total) return;
    int numKt = KP >> 5;
    int perNt = numKt << 9;                 // numKt * 512
    int nt = idx / perNt;
    int r  = idx - nt * perNt;
    int kt = r >> 9;
    int r2 = r & 511;
    int lane = r2 >> 4;
    int j    = r2 & 15;
    int l  = lane & 15;
    int hi = lane >> 4;
    int n = nt * 16 + l;
    int k = kt * 32 + hi * 8 + (j & 7) + (j >> 3) * 16;
    float v = (k < K) ? src[k * NCOL + n] : 0.0f;
    dst[idx] = (_Float16)v;
}

__global__ void zero_kernel(float* __restrict__ p, unsigned long long n) {
    unsigned long long i = (unsigned long long)blockIdx.x * blockDim.x + threadIdx.x;
    unsigned long long stride = (unsigned long long)gridDim.x * blockDim.x;
    for (; i < n; i += stride) p[i] = 0.0f;
}

// ---------------------------------------------------------------------------
// Async global->LDS chunk copy (CDNA5 GLOBAL_LOAD_ASYNC_TO_LDS, ASYNCcnt).
// Copies `halves` f16 (multiple of 256) from g to lds; 512B per wave-issue.
// ---------------------------------------------------------------------------
__device__ __forceinline__
void async_copy_chunk(const _Float16* __restrict__ g, _Float16* lds,
                      int halves, int lane) {
#if USE_ASYNC
    for (int i = lane; i * 8 < halves; i += 32) {
        unsigned long long ga = (unsigned long long)(g + i * 8);
        unsigned lo = (unsigned)(unsigned long long)(lds + i * 8);
        asm volatile("global_load_async_to_lds_b128 %0, %1, off"
                     :: "v"(lo), "v"(ga) : "memory");
    }
#else
    typedef __attribute__((ext_vector_type(8))) _Float16 v8h;
    for (int i = lane; i * 8 < halves; i += 32)
        *(v8h*)(lds + i * 8) = *(const v8h*)(g + i * 8);
#endif
}

__device__ __forceinline__ void async_wait_le(int) {}  // doc helper (unused)

#if USE_ASYNC
#define ASYNC_WAIT(n) asm volatile("s_wait_asynccnt %0" :: "i"(n) : "memory")
#else
#define ASYNC_WAIT(n) __builtin_amdgcn_wave_barrier()
#endif

// ---------------------------------------------------------------------------
// Fused PointConv edge-MLP layer (one wave per block, MT=2 edge-tiles):
//   msg = [x[src], pos[src]-pos[dst]]  (DIN = DIN_X + 2, zero-padded to DIN_PAD)
//   h   = relu(msg @ Wa + ba) @ Wb + bb
//   agg[dst] = max(agg[dst], relu(h))   via int-punned atomicMax (agg zero-init)
// A/H tiles live in LDS in native WMMA fragment layout. B fragments are
// streamed global->LDS with double-buffered async copies overlapping WMMA.
// ---------------------------------------------------------------------------
template<int DIN_X, int DIN_PAD, int H, int DOUT>
__global__ __launch_bounds__(32)
void edge_mlp_kernel(const float* __restrict__ x,
                     const float* __restrict__ pos,
                     const int*   __restrict__ src,
                     const int*   __restrict__ dst,
                     const _Float16* __restrict__ WaP,
                     const float* __restrict__ ba,
                     const _Float16* __restrict__ WbP,
                     const float* __restrict__ bb,
                     float* __restrict__ agg,
                     int E_, int numTiles) {
    constexpr int MT  = 2;
    constexpr int KT1 = DIN_PAD / 32, NT1 = H / 16;
    constexpr int KT2 = H / 32,       NT2 = DOUT / 16;
    constexpr int CH1 = KT1 * 512,    CH2 = KT2 * 512;   // halves per nt-chunk
    constexpr int ISS1 = CH1 / 256,   ISS2 = CH2 / 256;  // async issues per chunk

    __shared__ alignas(32) _Float16 sA[MT * 16 * DIN_PAD];
    __shared__ alignas(32) _Float16 sH[MT * 16 * H];
    __shared__ alignas(32) _Float16 sB[2][KT2 * 512];    // KT2 >= KT1 for all layers
    __shared__ int   sSrcS[MT * 16];
    __shared__ int   sDstS[MT * 16];
    __shared__ float sRel[MT * 16 * 2];

    const int lane  = threadIdx.x & 31;
    const int l15   = lane & 15;
    const int Mbase = (lane >> 4) * 8;

    int  tiles[MT];
    bool tv[MT];
#pragma unroll
    for (int mt = 0; mt < MT; ++mt) {
        tiles[mt] = blockIdx.x * MT + mt;
        tv[mt] = tiles[mt] < numTiles;
    }

    // ---- stage edge metadata (src, dst, rel-pos) -------------------------
#pragma unroll
    for (int mt = 0; mt < MT; ++mt) {
        if (lane < 16) {
            int e = tiles[mt] * 16 + lane;
            if (e >= E_) e = E_ - 1;
            int s = src[e], d = dst[e];
            sSrcS[mt * 16 + lane] = s;
            sDstS[mt * 16 + lane] = d;
            if (DIN_X > 0) __builtin_prefetch((const void*)(x + (long long)s * DIN_X), 0, 1);
            sRel[(mt * 16 + lane) * 2 + 0] = pos[2 * s + 0] - pos[2 * d + 0];
            sRel[(mt * 16 + lane) * 2 + 1] = pos[2 * s + 1] - pos[2 * d + 1];
        }
    }
    __builtin_amdgcn_wave_barrier();

    // ---- prefetch first B chunk of GEMM1 asynchronously ------------------
    async_copy_chunk(WaP, &sB[0][0], CH1, lane);

    // ---- build A tiles in WMMA A-fragment layout -------------------------
#pragma unroll
    for (int mt = 0; mt < MT; ++mt) {
        _Float16* pa = sA + mt * 16 * DIN_PAD;
        for (int idx = lane; idx < 16 * DIN_PAD; idx += 32) {
            int m = idx / DIN_PAD;
            int k = idx - m * DIN_PAD;
            float v = 0.0f;
            if (k < DIN_X) {
                int s = sSrcS[mt * 16 + m];
                v = x[(long long)s * DIN_X + k];
            } else if (k == DIN_X) {
                v = sRel[(mt * 16 + m) * 2 + 0];
            } else if (k == DIN_X + 1) {
                v = sRel[(mt * 16 + m) * 2 + 1];
            }
            int kt = k >> 5, r = k & 31;
            int hA = (r >> 3) & 1;
            int j  = (r & 7) + ((r >> 4) & 1) * 8;
            pa[kt * 512 + (m + 16 * hA) * 16 + j] = (_Float16)v;
        }
    }
    __builtin_amdgcn_wave_barrier();

    // ---- GEMM1: [16 x DIN_PAD] @ [DIN_PAD x H], bias + relu -> sH --------
    for (int nt = 0; nt < NT1; ++nt) {
        const int buf = nt & 1;
        if (nt + 1 < NT1) {
            async_copy_chunk(WaP + (nt + 1) * CH1, &sB[(nt + 1) & 1][0], CH1, lane);
            ASYNC_WAIT(ISS1);           // chunk nt complete, chunk nt+1 in flight
        } else {
            ASYNC_WAIT(0);
        }
        __builtin_amdgcn_wave_barrier();

        float bias = ba[nt * 16 + l15];
        v8f acc[MT];
#pragma unroll
        for (int mt = 0; mt < MT; ++mt)
#pragma unroll
            for (int r = 0; r < 8; ++r) acc[mt][r] = bias;

        for (int kt = 0; kt < KT1; ++kt) {
            const v16h b = *(const v16h*)(&sB[buf][0] + (kt << 9) + (lane << 4));
#pragma unroll
            for (int mt = 0; mt < MT; ++mt) {
                const v16h a = *(const v16h*)(sA + mt * 16 * DIN_PAD
                                              + (kt << 9) + (lane << 4));
                acc[mt] = __builtin_amdgcn_wmma_f32_16x16x32_f16(
                    false, a, false, b, (short)0, acc[mt], false, false);
            }
        }
        // relu + scatter into sH in A-fragment layout (C/D layout -> A layout)
        int colN = nt * 16 + l15;
        int kt2 = colN >> 5, r2 = colN & 31;
        int hA = (r2 >> 3) & 1;
        int j  = (r2 & 7) + ((r2 >> 4) & 1) * 8;
#pragma unroll
        for (int mt = 0; mt < MT; ++mt) {
            _Float16* ph = sH + mt * 16 * H;
#pragma unroll
            for (int r = 0; r < 8; ++r) {
                int M = Mbase + r;
                float v = fmaxf(acc[mt][r], 0.0f);
                ph[kt2 * 512 + (M + 16 * hA) * 16 + j] = (_Float16)v;
            }
        }
        __builtin_amdgcn_wave_barrier();
    }

    // ---- prefetch first B chunk of GEMM2 ---------------------------------
    async_copy_chunk(WbP, &sB[0][0], CH2, lane);

    // ---- GEMM2: [16 x H] @ [H x DOUT], bias, relu, atomic scatter-max ----
    for (int nt = 0; nt < NT2; ++nt) {
        const int buf = nt & 1;
        if (nt + 1 < NT2) {
            async_copy_chunk(WbP + (nt + 1) * CH2, &sB[(nt + 1) & 1][0], CH2, lane);
            ASYNC_WAIT(ISS2);
        } else {
            ASYNC_WAIT(0);
        }
        __builtin_amdgcn_wave_barrier();

        float bias = bb[nt * 16 + l15];
        v8f acc[MT];
#pragma unroll
        for (int mt = 0; mt < MT; ++mt)
#pragma unroll
            for (int r = 0; r < 8; ++r) acc[mt][r] = bias;

        for (int kt = 0; kt < KT2; ++kt) {
            const v16h b = *(const v16h*)(&sB[buf][0] + (kt << 9) + (lane << 4));
#pragma unroll
            for (int mt = 0; mt < MT; ++mt) {
                const v16h a = *(const v16h*)(sH + mt * 16 * H
                                              + (kt << 9) + (lane << 4));
                acc[mt] = __builtin_amdgcn_wmma_f32_16x16x32_f16(
                    false, a, false, b, (short)0, acc[mt], false, false);
            }
        }
        int col = nt * 16 + l15;
#pragma unroll
        for (int mt = 0; mt < MT; ++mt) {
            if (!tv[mt]) continue;
#pragma unroll
            for (int r = 0; r < 8; ++r) {
                int M = Mbase + r;
                int eIdx = tiles[mt] * 16 + M;
                if (eIdx < E_) {
                    int node = sDstS[mt * 16 + M];
                    float v = fmaxf(acc[mt][r], 0.0f);
                    atomicMax((int*)agg + (long long)node * DOUT + col,
                              __float_as_int(v));
                }
            }
        }
        __builtin_amdgcn_wave_barrier();
    }
}

// ---------------------------------------------------------------------------
// Global max pool over nodes: pooled[c] = max_n agg3[n*256+c] (all values >=0)
// ---------------------------------------------------------------------------
__global__ void pool_kernel(const float* __restrict__ agg3,
                            float* __restrict__ pooled, int n) {
    int col = threadIdx.x;                  // 256 threads
    int chunk = (n + gridDim.x - 1) / gridDim.x;
    int n0 = blockIdx.x * chunk;
    int n1 = n0 + chunk; if (n1 > n) n1 = n;
    float m = 0.0f;
    for (int i = n0; i < n1; ++i)
        m = fmaxf(m, agg3[(long long)i * 256 + col]);
    atomicMax((int*)pooled + col, __float_as_int(m));
}

// ---------------------------------------------------------------------------
// Dense head: relu(x@Wl1+bl1) -> relu(@Wl2+bl2) -> @Wl3+bl3  (x is [1,256])
// ---------------------------------------------------------------------------
__global__ void head_kernel(const float* __restrict__ pooled,
                            const float* __restrict__ Wl1, const float* __restrict__ bl1,
                            const float* __restrict__ Wl2, const float* __restrict__ bl2,
                            const float* __restrict__ Wl3, const float* __restrict__ bl3,
                            float* __restrict__ out) {
    __shared__ float sP[256];
    __shared__ float sY[256];
    __shared__ float sR[256];
    int t = threadIdx.x;
    sP[t] = pooled[t];
    __syncthreads();

    float acc = bl1[t];
    for (int i = 0; i < 256; ++i) acc += sP[i] * Wl1[i * 256 + t];
    acc = fmaxf(acc, 0.0f);
    sY[t] = acc;
    __syncthreads();

    acc = bl2[t];
    for (int i = 0; i < 256; ++i) acc += sY[i] * Wl2[i * 256 + t];
    acc = fmaxf(acc, 0.0f);
    __syncthreads();
    sP[t] = acc;
    __syncthreads();

    sR[t] = sP[t] * Wl3[t];
    __syncthreads();
    for (int s = 128; s > 0; s >>= 1) {
        if (t < s) sR[t] += sR[t + s];
        __syncthreads();
    }
    if (t == 0) out[0] = sR[0] + bl3[0];
}

// ---------------------------------------------------------------------------
extern "C" void kernel_launch(void* const* d_in, const int* in_sizes, int n_in,
                              void* d_out, int out_size, void* d_ws, size_t ws_size,
                              hipStream_t stream) {
    (void)n_in; (void)out_size; (void)ws_size;
    const float* pos = (const float*)d_in[0];
    const int*   ei  = (const int*)d_in[1];
    const int Nn = in_sizes[0] / 2;
    const int Ee = in_sizes[1] / 2;
    const int* src = ei;
    const int* dst = ei + Ee;

    const float* W1a = (const float*)d_in[2];  const float* b1a = (const float*)d_in[3];
    const float* W1b = (const float*)d_in[4];  const float* b1b = (const float*)d_in[5];
    const float* W2a = (const float*)d_in[6];  const float* b2a = (const float*)d_in[7];
    const float* W2b = (const float*)d_in[8];  const float* b2b = (const float*)d_in[9];
    const float* W3a = (const float*)d_in[10]; const float* b3a = (const float*)d_in[11];
    const float* W3b = (const float*)d_in[12]; const float* b3b = (const float*)d_in[13];
    const float* Wl1 = (const float*)d_in[14]; const float* bl1 = (const float*)d_in[15];
    const float* Wl2 = (const float*)d_in[16]; const float* bl2 = (const float*)d_in[17];
    const float* Wl3 = (const float*)d_in[18]; const float* bl3 = (const float*)d_in[19];

    // ---- workspace layout ------------------------------------------------
    char* base = (char*)d_ws;
    size_t off = 0;
    auto take = [&](size_t bytes) -> void* {
        void* p = base + off;
        off += (bytes + 255) & ~(size_t)255;
        return p;
    };
    _Float16* w1a = (_Float16*)take((size_t)32  * 64  * 2);
    _Float16* w1b = (_Float16*)take((size_t)64  * 64  * 2);
    _Float16* w2a = (_Float16*)take((size_t)96  * 128 * 2);
    _Float16* w2b = (_Float16*)take((size_t)128 * 128 * 2);
    _Float16* w3a = (_Float16*)take((size_t)160 * 256 * 2);
    _Float16* w3b = (_Float16*)take((size_t)256 * 256 * 2);
    size_t ftotal = (size_t)Nn * (64 + 128 + 256) + 256;
    float* fbase  = (float*)take(ftotal * 4);
    float* agg1   = fbase;
    float* agg2   = agg1 + (size_t)Nn * 64;
    float* agg3   = agg2 + (size_t)Nn * 128;
    float* pooled = agg3 + (size_t)Nn * 256;

    // ---- pack weights to f16 WMMA fragment order -------------------------
    auto packLaunch = [&](const float* s, _Float16* d, int K, int NC, int KP) {
        int tot = KP * NC;
        pack_weights_kernel<<<(tot + 255) / 256, 256, 0, stream>>>(s, d, K, NC, KP);
    };
    packLaunch(W1a, w1a, 2,   64,  32);
    packLaunch(W1b, w1b, 64,  64,  64);
    packLaunch(W2a, w2a, 66,  128, 96);
    packLaunch(W2b, w2b, 128, 128, 128);
    packLaunch(W3a, w3a, 130, 256, 160);
    packLaunch(W3b, w3b, 256, 256, 256);

    // ---- zero aggregation buffers + pooled vector ------------------------
    zero_kernel<<<2048, 256, 0, stream>>>(fbase, (unsigned long long)ftotal);

    const int numTiles = (Ee + 15) / 16;
    const int grid = (numTiles + 1) / 2;   // 1 wave x 2 tiles per block

    edge_mlp_kernel<0, 32, 64, 64><<<grid, 32, 0, stream>>>(
        (const float*)nullptr, pos, src, dst, w1a, b1a, w1b, b1b, agg1, Ee, numTiles);
    edge_mlp_kernel<64, 96, 128, 128><<<grid, 32, 0, stream>>>(
        agg1, pos, src, dst, w2a, b2a, w2b, b2b, agg2, Ee, numTiles);
    edge_mlp_kernel<128, 160, 256, 256><<<grid, 32, 0, stream>>>(
        agg2, pos, src, dst, w3a, b3a, w3b, b3b, agg3, Ee, numTiles);

    pool_kernel<<<128, 256, 0, stream>>>(agg3, pooled, Nn);
    head_kernel<<<1, 256, 0, stream>>>(pooled, Wl1, bl1, Wl2, bl2, Wl3, bl3,
                                       (float*)d_out);
}
